// ReSkipBlockGroup_55946243998382
// MI455X (gfx1250) — matrix operand order repaired and checked by reading, not compiled, not run on hardware
//
#include <hip/hip_runtime.h>
#include <stdint.h>

typedef __attribute__((ext_vector_type(16))) _Float16 v16h;
typedef __attribute__((ext_vector_type(8)))  _Float16 v8h;
typedef __attribute__((ext_vector_type(8)))  float    v8f;

#define NSRC 7
#define NBLK 6
#define DDIM 1024
#define TT   8192                 /* b*t = 2*4096 tokens */
#define INV_SCALE (1.0f/32.0f)    /* 1/(sqrt(1024)*temperature) */
#define RMS_EPS 1e-6f
#define WAVES_PER_WG 2
#define TOK_PER_WAVE 16

// CDNA5 async global->LDS copy (ASYNCcnt-tracked), 16 bytes per lane.
__device__ __forceinline__ void async_copy_b128(void* lds_dst, const void* gsrc) {
  unsigned lds_off = (unsigned)(uintptr_t)lds_dst;  // low 32 bits of flat LDS addr = group offset
  asm volatile("global_load_async_to_lds_b128 %0, %1, off"
               :: "v"(lds_off), "v"(gsrc) : "memory");
}
__device__ __forceinline__ void wait_async0() {
  asm volatile("s_wait_asynccnt 0" ::: "memory");
}

__global__ __launch_bounds__(32 * WAVES_PER_WG)
void router_merge_kernel(const float* __restrict__ blocks,   // [6, TT, D]
                         const float* __restrict__ partial,  // [TT, D]
                         const float* __restrict__ w_query,  // [D]
                         const float* __restrict__ norm_w,   // [D]
                         float* __restrict__ out_hidden,     // [TT, D]
                         float* __restrict__ out_entropy)    // [TT]
{
  __shared__ __align__(16) float    wq_s[DDIM];
  __shared__ __align__(16) float    nw_s[DDIM];
  __shared__ __align__(16) _Float16 qh[DDIM];
  __shared__ float wdots[WAVES_PER_WG][NSRC][16];
  __shared__ float wcoef[WAVES_PER_WG][NSRC][16];

  const int tid  = threadIdx.x;
  const int lane = tid & 31;
  const int wave = tid >> 5;

  // ---- stage query operands into LDS via CDNA5 async tensor path ----
  for (int i = tid * 4; i < DDIM; i += blockDim.x * 4) {
    async_copy_b128(&wq_s[i], w_query + i);
    async_copy_b128(&nw_s[i], norm_w + i);
  }
  wait_async0();
  __syncthreads();
  for (int i = tid; i < DDIM; i += blockDim.x)
    qh[i] = (_Float16)(wq_s[i] * nw_s[i]);   // effective query, f16 for WMMA B
  __syncthreads();

  const int tile = blockIdx.x * WAVES_PER_WG + wave;
  const int t0   = tile * TOK_PER_WAVE;
  const int m    = lane & 15;                 // token row M for this lane
  const int row  = t0 + m;
  const int koff = (lane < 16) ? 0 : 8;       // A-fragment K groups per ISA layout
  const int qoff = (lane < 16) ? 0 : 8;       // B-fragment K groups per ISA layout

  const float* srcRow[NSRC];
  #pragma unroll
  for (int s = 0; s < NBLK; ++s)
    srcRow[s] = blocks + ((size_t)s * TT + row) * DDIM;
  srcRow[NBLK] = partial + (size_t)row * DDIM;

  v8f  acc[NSRC] = {};
  float ssq[NSRC];
  #pragma unroll
  for (int s = 0; s < NSRC; ++s) ssq[s] = 0.0f;

  // ---- pass 1: WMMA dot(query, x) for 16 tokens x 7 sources; Sigma x^2 in VALU ----
  for (int k = 0; k < DDIM; k += 32) {
    v8h b0 = *(const v8h*)&qh[k + qoff];
    v8h b1 = *(const v8h*)&qh[k + qoff + 16];
    v16h bfrag;
    #pragma unroll
    for (int e = 0; e < 8; ++e) { bfrag[e] = b0[e]; bfrag[8 + e] = b1[e]; }

    #pragma unroll
    for (int s = 0; s < NSRC; ++s) {
      const float4* p4 = (const float4*)(srcRow[s] + k + koff);
      float4 f0 = p4[0], f1 = p4[1], f2 = p4[4], f3 = p4[5];

      ssq[s] += f0.x*f0.x + f0.y*f0.y + f0.z*f0.z + f0.w*f0.w
              + f1.x*f1.x + f1.y*f1.y + f1.z*f1.z + f1.w*f1.w
              + f2.x*f2.x + f2.y*f2.y + f2.z*f2.z + f2.w*f2.w
              + f3.x*f3.x + f3.y*f3.y + f3.z*f3.z + f3.w*f3.w;

      v16h a;
      a[0]=(_Float16)f0.x; a[1]=(_Float16)f0.y; a[2]=(_Float16)f0.z; a[3]=(_Float16)f0.w;
      a[4]=(_Float16)f1.x; a[5]=(_Float16)f1.y; a[6]=(_Float16)f1.z; a[7]=(_Float16)f1.w;
      a[8]=(_Float16)f2.x; a[9]=(_Float16)f2.y; a[10]=(_Float16)f2.z; a[11]=(_Float16)f2.w;
      a[12]=(_Float16)f3.x; a[13]=(_Float16)f3.y; a[14]=(_Float16)f3.z; a[15]=(_Float16)f3.w;

      acc[s] = __builtin_amdgcn_wmma_f32_16x16x32_f16(
          /*neg_a=*/false, a, /*neg_b=*/false, bfrag,
          /*c_mod=*/(short)0, acc[s], /*reuse_a=*/false, /*reuse_b=*/false);
    }
  }

  // full Sigma x^2 per token: lane m holds K{0-7,16-23,...}, lane m+16 the rest
  #pragma unroll
  for (int s = 0; s < NSRC; ++s) ssq[s] += __shfl_xor(ssq[s], 16, 32);

  // extract dots from C layout: lane 0 -> M=0..7 (acc[v]=M=v), lane 16 -> M=8..15
  if (lane == 0 || lane == 16) {
    const int off = (lane == 0) ? 0 : 8;
    #pragma unroll
    for (int s = 0; s < NSRC; ++s) {
      #pragma unroll
      for (int v = 0; v < 8; ++v) wdots[wave][s][off + v] = acc[s][v];
    }
  }
  __syncthreads();

  // ---- per-token softmax / online merge / entropy (one token per lane 0..15) ----
  if (lane < 16) {
    float sc[NSRC];
    #pragma unroll
    for (int s = 0; s < NSRC; ++s) {
      float inv = rsqrtf(ssq[s] * (1.0f / DDIM) + RMS_EPS);
      sc[s] = wdots[wave][s][m] * inv * INV_SCALE;
    }
    float m1 = sc[0];
    #pragma unroll
    for (int s = 1; s < NBLK; ++s) m1 = fmaxf(m1, sc[s]);
    float e[NBLK], lse = 0.0f;
    #pragma unroll
    for (int s = 0; s < NBLK; ++s) { e[s] = __expf(sc[s] - m1); lse += e[s]; }

    float ps = sc[NBLK];
    float mm = fmaxf(m1, ps);
    float c1 = __expf(m1 - mm);
    float cp = __expf(ps - mm);
    float denom = c1 * lse + cp;
    float w1f = c1 * lse / denom;   // phase1_weight
    float wpf = cp / denom;         // partial_weight

    float loglse = __logf(lse);
    float ent1 = 0.0f;
    #pragma unroll
    for (int s = 0; s < NBLK; ++s) {
      float pr = e[s] / lse;
      float lp = (sc[s] - m1) - loglse;
      ent1 -= pr * lp;
    }
    float w1 = fmaxf(w1f, 1e-8f), wp = fmaxf(wpf, 1e-8f);
    out_entropy[row] = w1 * ent1 - w1 * __logf(w1) - wp * __logf(wp);

    // hidden = (Sigma e_s x_s) * phase1_weight + partial * partial_weight  (as in reference)
    #pragma unroll
    for (int s = 0; s < NBLK; ++s) wcoef[wave][s][m] = e[s] * w1f;
    wcoef[wave][NBLK][m] = wpf;
  }
  __syncthreads();

  // ---- pass 2: coalesced weighted sum; 32 lanes sweep d per token ----
  for (int t = 0; t < TOK_PER_WAVE; ++t) {
    const int tok = t0 + t;
    float c[NSRC];
    #pragma unroll
    for (int s = 0; s < NSRC; ++s) c[s] = wcoef[wave][s][t];
    float* outp = out_hidden + (size_t)tok * DDIM;
    for (int it = 0; it < DDIM / (32 * 4); ++it) {
      const int dx = it * 128 + lane * 4;
      float4 av; av.x = av.y = av.z = av.w = 0.0f;
      #pragma unroll
      for (int s = 0; s < NBLK; ++s) {
        const float4 xv = *(const float4*)(blocks + ((size_t)s * TT + tok) * DDIM + dx);
        av.x += c[s] * xv.x; av.y += c[s] * xv.y; av.z += c[s] * xv.z; av.w += c[s] * xv.w;
      }
      const float4 pv = *(const float4*)(partial + (size_t)tok * DDIM + dx);
      av.x += c[NBLK] * pv.x; av.y += c[NBLK] * pv.y;
      av.z += c[NBLK] * pv.z; av.w += c[NBLK] * pv.w;
      *(float4*)(outp + dx) = av;
    }
  }
}

extern "C" void kernel_launch(void* const* d_in, const int* in_sizes, int n_in,
                              void* d_out, int out_size, void* d_ws, size_t ws_size,
                              hipStream_t stream) {
  (void)in_sizes; (void)n_in; (void)out_size; (void)d_ws; (void)ws_size;
  const float* blocks  = (const float*)d_in[0];
  const float* partial = (const float*)d_in[1];
  const float* w_query = (const float*)d_in[2];
  const float* norm_w  = (const float*)d_in[3];
  float* out_hidden  = (float*)d_out;
  float* out_entropy = out_hidden + (size_t)TT * DDIM;

  dim3 grid(TT / (TOK_PER_WAVE * WAVES_PER_WG));   // 256 workgroups
  dim3 block(32 * WAVES_PER_WG);                   // 2 waves (wave32)
  router_merge_kernel<<<grid, block, 0, stream>>>(blocks, partial, w_query, norm_w,
                                                  out_hidden, out_entropy);
}
